// RelationMPGNN_19516331393713
// MI455X (gfx1250) — matrix operand order repaired and checked by reading, not compiled, not run on hardware
//
#include <hip/hip_runtime.h>
#include <hip/hip_bf16.h>

// RelationMPGNN on gfx1250: all GEMMs via v_wmma_f32_16x16x32_f16 (wave32),
// wave = 16x32 output tile (1 A fragment -> 2 WMMAs), fused edge gathers in
// the A-fragment address selection, fused segment-sum via
// global_atomic_add_f32 in the epilogue. All A chunks are unconditional
// 16B loads from cndmask-selected addresses (zero-pad regions are backed by
// real zero-filled memory), so no scalar fallback paths exist.

typedef _Float16 h16_t;
typedef __attribute__((ext_vector_type(16))) _Float16 v16h;
typedef __attribute__((ext_vector_type(8)))  _Float16 v8h;
typedef __attribute__((ext_vector_type(8)))  float    v8f;

#define NN   20000
#define EE   320000
#define QQ   100000
#define HH   256
#define LL   3

// ---------------------------------------------------------------- utilities

__global__ void k_cast_f2h(const float* __restrict__ s, h16_t* __restrict__ d, long n) {
  long i = (long)blockIdx.x * blockDim.x + threadIdx.x;
  if (i < n) d[i] = (h16_t)s[i];
}

__global__ void k_zero_f32(float* __restrict__ p, long n) {
  long i = (long)blockIdx.x * blockDim.x + threadIdx.x;
  if (i < n) p[i] = 0.f;
}

// geom_x [N,18] fp32 -> [N,32] fp16, zero padded (keeps 16B-aligned K chunks)
__global__ void k_geom_pad(const float* __restrict__ g, h16_t* __restrict__ d, int N) {
  long i = (long)blockIdx.x * blockDim.x + threadIdx.x;
  long total = (long)N * 32;
  if (i >= total) return;
  int  c = (int)(i & 31);
  long n = i >> 5;
  float v = (c < 18) ? g[n * 18 + c] : 0.f;
  d[i] = (h16_t)v;
}

// Pack fp32 weight [Kreal,Nreal] row-major into WMMA B-fragment order:
// dst[(((nt*nKt)+kt)*32 + lane)*16 + i] = W[kt*32 + (i>>3)*16 + (lane>>4)*8 + (i&7)][nt*16 + (lane&15)]
__global__ void k_pack_w(const float* __restrict__ src, h16_t* __restrict__ dst,
                         int Kreal, int Nreal, int Kp, int Np) {
  long t = (long)blockIdx.x * blockDim.x + threadIdx.x;
  int  nKt = Kp >> 5;
  long total = (long)(Np >> 4) * nKt * 512;
  if (t >= total) return;
  int  i    = (int)(t & 15);
  int  lane = (int)((t >> 4) & 31);
  long frag = t >> 9;
  int  kt   = (int)(frag % nKt);
  int  nt   = (int)(frag / nKt);
  int  k = kt * 32 + ((i >> 3) << 4) + ((lane >> 4) << 3) + (i & 7);
  int  n = nt * 16 + (lane & 15);
  float v = (k < Kreal && n < Nreal) ? src[(long)k * Nreal + n] : 0.f;
  dst[t] = (h16_t)v;
}

// per-query 8 relational features -> fp16 [Q,32], zero padded
__global__ void k_rel(const float* __restrict__ geom, const int* __restrict__ qp,
                      h16_t* __restrict__ rel, int Q) {
  int q = blockIdx.x * blockDim.x + threadIdx.x;
  if (q >= Q) return;
  int s = qp[2 * q], o = qp[2 * q + 1];
  const float* gs = geom + (long)s * 18;
  const float* go = geom + (long)o * 18;
  float dx = go[0] - gs[0], dy = go[1] - gs[1], dz = go[2] - gs[2];
  float dist = sqrtf(dx * dx + dy * dy + dz * dz);
  float logd = log1pf(dist);
  float r0 = logf(fmaxf(go[3], 1e-6f) / fmaxf(gs[3], 1e-6f));
  float r1 = logf(fmaxf(go[4], 1e-6f) / fmaxf(gs[4], 1e-6f));
  float r2 = logf(fmaxf(go[5], 1e-6f) / fmaxf(gs[5], 1e-6f));
  float sx = gs[15], sy = gs[16], sz = gs[17];
  float ox = go[15], oy = go[16], oz = go[17];
  float ns = sqrtf(sx * sx + sy * sy + sz * sz);
  float no = sqrtf(ox * ox + oy * oy + oz * oz);
  float cosv = (sx * ox + sy * oy + sz * oz) / (fmaxf(ns, 1e-8f) * fmaxf(no, 1e-8f));
  h16_t* r = rel + (long)q * 32;
  r[0] = (h16_t)dx;   r[1] = (h16_t)dy;  r[2] = (h16_t)dz;  r[3] = (h16_t)logd;
  r[4] = (h16_t)r0;   r[5] = (h16_t)r1;  r[6] = (h16_t)r2;  r[7] = (h16_t)cosv;
  for (int i = 8; i < 32; ++i) r[i] = (h16_t)0.f;
}

// ---------------------------------------------------------------- WMMA GEMM

// A-source modes (all 8-half chunks are unconditional vector loads; all
// segment boundaries in every launch are multiples of 8):
//  0 PLAIN       : A0[m*L0+k]                          (L0 == Kp)
//  1 CONCAT2     : [A0 (stride L0) | A1 (stride L1)]   (L0+L1 == Kp)
//  2 GATHER2     : [A0[r0] | A0[r1]] each L0 wide      (Kp == 2*L0)
//  3 GATHER2_PAD : [A0[r0] | A0[r1] | A2[m] stride L1] (2*L0+L1 == Kp)
template<int AMODE>
__device__ __forceinline__ const h16_t* chunkPtr(int m, int k,
    const h16_t* __restrict__ A0, const h16_t* __restrict__ A1,
    const h16_t* __restrict__ A2, int L0, int L1, int r0, int r1) {
  if (AMODE == 0) return A0 + (size_t)m * L0 + k;
  if (AMODE == 1) return (k < L0) ? A0 + (size_t)m * L0 + k
                                  : A1 + (size_t)m * L1 + (k - L0);
  if (AMODE == 2) return (k < L0) ? A0 + (size_t)r0 * L0 + k
                                  : A0 + (size_t)r1 * L0 + (k - L0);
  return (k < L0)     ? A0 + (size_t)r0 * L0 + k
       : (k < 2 * L0) ? A0 + (size_t)r1 * L0 + (k - L0)
                      : A2 + (size_t)m * L1 + (k - 2 * L0);
}

// Epilogues:
//  0 relu -> O16        1 bias + atomicAdd(O32[segIdx[row]*strideO+col])
//  2 relu -> O32 & O16  3 residual: O32 += y, O16 = (h)O32
//  4 f32 store, col < ncolValid only (no relu)
template<int AMODE, int EPI>
__global__ void __launch_bounds__(256) k_gemm(
    const h16_t* __restrict__ A0, const h16_t* __restrict__ A1,
    const h16_t* __restrict__ A2,
    const int* __restrict__ gidx0, const int* __restrict__ gidx1, int gstride,
    int L0, int L1,
    const h16_t* __restrict__ Wp, const float* __restrict__ bias,
    int M, int N, int Kp,
    float* __restrict__ O32, h16_t* __restrict__ O16,
    const int* __restrict__ segIdx, int segStride,
    int strideO, int ncolValid) {
  const int  lane   = threadIdx.x & 31;
  const long tile   = (long)blockIdx.x * 8 + (threadIdx.x >> 5);
  const int  ntN2   = N >> 5;                 // wave computes 16x32 (2 n-tiles)
  const long nTiles = (long)(M >> 4) * ntN2;
  if (tile >= nTiles) return;                 // wave-uniform exit
  const int tile_n2 = (int)(tile % ntN2);
  const int tile_m  = (int)(tile / ntN2);

  const int mrow = tile_m * 16 + (lane & 15); // A row this lane feeds
  const int kb   = (lane >> 4) << 3;          // A chunk base for lane half

  int r0 = 0, r1 = 0;
  if (AMODE >= 2) {
    r0 = gidx0[(size_t)mrow * gstride];
    r1 = gidx1[(size_t)mrow * gstride];
  }

  v8f acc0 = {}, acc1 = {};
  const int nKt = Kp >> 5;
  const h16_t* wbase0 = Wp + (size_t)(tile_n2 * 2) * nKt * 512 + (size_t)lane * 16;
  const h16_t* wbase1 = wbase0 + (size_t)nKt * 512;

  for (int k0 = 0; k0 < Kp; k0 += 32) {
    v8h lo = *(const v8h*)chunkPtr<AMODE>(mrow, k0 + kb,      A0, A1, A2, L0, L1, r0, r1);
    v8h hi = *(const v8h*)chunkPtr<AMODE>(mrow, k0 + 16 + kb, A0, A1, A2, L0, L1, r0, r1);
    v16h a;
    for (int i = 0; i < 8; ++i) { a[i] = lo[i]; a[i + 8] = hi[i]; }
    const size_t ko = (size_t)(k0 >> 5) * 512;
    if (k0 + 32 < Kp) {
      __builtin_prefetch(wbase0 + ko + 512, 0, 1);
      __builtin_prefetch(wbase1 + ko + 512, 0, 1);
    }
    v16h b0 = *(const v16h*)(wbase0 + ko);
    v16h b1 = *(const v16h*)(wbase1 + ko);
    acc0 = __builtin_amdgcn_wmma_f32_16x16x32_f16(false, a, false, b0,
                                                  (short)0, acc0, false, false);
    acc1 = __builtin_amdgcn_wmma_f32_16x16x32_f16(false, a, false, b1,
                                                  (short)0, acc1, false, false);
  }

  const int rb = (lane >> 4) * 8;
#pragma unroll
  for (int t = 0; t < 2; ++t) {
    const v8f acc = t ? acc1 : acc0;
    const int col = (tile_n2 * 2 + t) * 16 + (lane & 15);
    const float bv = (col < ncolValid) ? bias[col] : 0.f;
    for (int r = 0; r < 8; ++r) {
      const int row = tile_m * 16 + rb + r;
      float y = acc[r] + bv;
      if (EPI == 0) {
        y = fmaxf(y, 0.f);
        O16[(size_t)row * strideO + col] = (h16_t)y;
      } else if (EPI == 1) {
        int seg = segIdx[(size_t)row * segStride];
        atomicAdd(O32 + (size_t)seg * strideO + col, y);
      } else if (EPI == 2) {
        y = fmaxf(y, 0.f);
        size_t o = (size_t)row * strideO + col;
        O32[o] = y; O16[o] = (h16_t)y;
      } else if (EPI == 3) {
        size_t o = (size_t)row * strideO + col;
        float nh = O32[o] + y;
        O32[o] = nh; O16[o] = (h16_t)nh;
      } else {
        if (col < ncolValid) O32[(size_t)row * strideO + col] = y;
      }
    }
  }
}

// ---------------------------------------------------------------- launcher

extern "C" void kernel_launch(void* const* d_in, const int* in_sizes, int n_in,
                              void* d_out, int out_size, void* d_ws, size_t ws_size,
                              hipStream_t stream) {
  (void)in_sizes; (void)n_in; (void)out_size; (void)ws_size;
  const float* clip_x  = (const float*)d_in[0];
  const float* geom_x  = (const float*)d_in[1];
  const int*   edges   = (const int*)d_in[2];
  const int*   qp      = (const int*)d_in[3];
  const float* clip_w1 = (const float*)d_in[4];
  const float* clip_b1 = (const float*)d_in[5];
  const float* clip_w2 = (const float*)d_in[6];
  const float* clip_b2 = (const float*)d_in[7];
  const float* fuse_w1 = (const float*)d_in[8];
  const float* fuse_b1 = (const float*)d_in[9];
  const float* fuse_w2 = (const float*)d_in[10];
  const float* fuse_b2 = (const float*)d_in[11];
  const float* msg_w1  = (const float*)d_in[12];
  const float* msg_b1  = (const float*)d_in[13];
  const float* msg_w2  = (const float*)d_in[14];
  const float* msg_b2  = (const float*)d_in[15];
  const float* upd_w1  = (const float*)d_in[16];
  const float* upd_b1  = (const float*)d_in[17];
  const float* upd_w2  = (const float*)d_in[18];
  const float* upd_b2  = (const float*)d_in[19];
  const float* head_w1 = (const float*)d_in[20];
  const float* head_b1 = (const float*)d_in[21];
  const float* head_w2 = (const float*)d_in[22];
  const float* head_b2 = (const float*)d_in[23];
  float* out = (float*)d_out;

  char* ws = (char*)d_ws;
  size_t off = 0;
  auto take = [&](size_t bytes) -> char* {
    char* p = ws + off;
    off = (off + bytes + 255) & ~(size_t)255;
    return p;
  };
  h16_t* x16   = (h16_t*)take((size_t)NN * 512 * 2);
  h16_t* z1    = (h16_t*)take((size_t)NN * 512 * 2);   // also reused as tmp[N,256]
  h16_t* z2    = (h16_t*)take((size_t)NN * 256 * 2);
  h16_t* geo   = (h16_t*)take((size_t)NN * 32 * 2);
  float* h32   = (float*)take((size_t)NN * 256 * 4);
  h16_t* h16v  = (h16_t*)take((size_t)NN * 256 * 2);
  float* agg32 = (float*)take((size_t)NN * 256 * 4);
  h16_t* agg16 = (h16_t*)take((size_t)NN * 256 * 2);
  h16_t* mmid  = (h16_t*)take((size_t)EE * 256 * 2);
  h16_t* rel   = (h16_t*)take((size_t)QQ * 32 * 2);
  h16_t* e16   = (h16_t*)take((size_t)QQ * 256 * 2);
  h16_t* pw_clip1 = (h16_t*)take((size_t)512 * 512 * 2);
  h16_t* pw_clip2 = (h16_t*)take((size_t)512 * 256 * 2);
  h16_t* pw_fuse1 = (h16_t*)take((size_t)288 * 256 * 2);
  h16_t* pw_fuse2 = (h16_t*)take((size_t)256 * 256 * 2);
  h16_t *pw_m1[LL], *pw_m2[LL], *pw_u1[LL], *pw_u2[LL];
  for (int l = 0; l < LL; ++l) {
    pw_m1[l] = (h16_t*)take((size_t)512 * 256 * 2);
    pw_m2[l] = (h16_t*)take((size_t)256 * 256 * 2);
    pw_u1[l] = (h16_t*)take((size_t)512 * 256 * 2);
    pw_u2[l] = (h16_t*)take((size_t)256 * 256 * 2);
  }
  h16_t* pw_h1 = (h16_t*)take((size_t)544 * 256 * 2);
  h16_t* pw_h2 = (h16_t*)take((size_t)256 * 32 * 2);

  auto cdiv = [](long a, long b) { return (a + b - 1) / b; };
  auto pack = [&](const float* src, h16_t* dst, int Kr, int Nr, int Kp, int Np) {
    long tot = (long)(Np / 16) * (Kp / 32) * 512;
    k_pack_w<<<(unsigned)cdiv(tot, 256), 256, 0, stream>>>(src, dst, Kr, Nr, Kp, Np);
  };
  auto grid = [&](long M, long N) { return dim3((unsigned)cdiv((M / 16) * (N / 32), 8)); };

  // --- input conversion + weight packing -----------------------------------
  k_cast_f2h<<<(unsigned)cdiv((long)NN * 512, 256), 256, 0, stream>>>(clip_x, x16, (long)NN * 512);
  k_geom_pad<<<(unsigned)cdiv((long)NN * 32, 256), 256, 0, stream>>>(geom_x, geo, NN);
  pack(clip_w1, pw_clip1, 512, 512, 512, 512);
  pack(clip_w2, pw_clip2, 512, 256, 512, 256);
  pack(fuse_w1, pw_fuse1, 274, 256, 288, 256);
  pack(fuse_w2, pw_fuse2, 256, 256, 256, 256);
  for (int l = 0; l < LL; ++l) {
    pack(msg_w1 + (size_t)l * 512 * 256, pw_m1[l], 512, 256, 512, 256);
    pack(msg_w2 + (size_t)l * 256 * 256, pw_m2[l], 256, 256, 256, 256);
    pack(upd_w1 + (size_t)l * 512 * 256, pw_u1[l], 512, 256, 512, 256);
    pack(upd_w2 + (size_t)l * 256 * 256, pw_u2[l], 256, 256, 256, 256);
  }
  pack(head_w1, pw_h1, 520, 256, 544, 256);
  pack(head_w2, pw_h2, 256, 26, 256, 32);

  // --- clip MLP: x -> z1 -> z2 ---------------------------------------------
  k_gemm<0, 0><<<grid(NN, 512), 256, 0, stream>>>(
      x16, nullptr, nullptr, nullptr, nullptr, 0, 512, 0,
      pw_clip1, clip_b1, NN, 512, 512, nullptr, z1, nullptr, 0, 512, 512);
  k_gemm<0, 0><<<grid(NN, 256), 256, 0, stream>>>(
      z1, nullptr, nullptr, nullptr, nullptr, 0, 512, 0,
      pw_clip2, clip_b2, NN, 256, 512, nullptr, z2, nullptr, 0, 256, 256);

  // --- fuse MLP: concat(z2, geom) -> h (f32 + f16) -------------------------
  k_gemm<1, 0><<<grid(NN, 256), 256, 0, stream>>>(
      z2, geo, nullptr, nullptr, nullptr, 0, 256, 32,
      pw_fuse1, fuse_b1, NN, 256, 288, nullptr, z1, nullptr, 0, 256, 256);
  k_gemm<0, 2><<<grid(NN, 256), 256, 0, stream>>>(
      z1, nullptr, nullptr, nullptr, nullptr, 0, 256, 0,
      pw_fuse2, fuse_b2, NN, 256, 256, h32, h16v, nullptr, 0, 256, 256);

  // --- message-passing layers ----------------------------------------------
  for (int l = 0; l < LL; ++l) {
    k_zero_f32<<<(unsigned)cdiv((long)NN * 256, 256), 256, 0, stream>>>(agg32, (long)NN * 256);
    // msg1: gather concat(h[src], h[dst]) @ W1 + b1, relu -> mmid (fp16)
    k_gemm<2, 0><<<grid(EE, 256), 256, 0, stream>>>(
        h16v, nullptr, nullptr, edges, edges + 1, 2, 256, 0,
        pw_m1[l], msg_b1 + (size_t)l * 256, EE, 256, 512,
        nullptr, mmid, nullptr, 0, 256, 256);
    // msg2: mmid @ W2 + b2, atomic segment-sum into agg32 by dst
    k_gemm<0, 1><<<grid(EE, 256), 256, 0, stream>>>(
        mmid, nullptr, nullptr, nullptr, nullptr, 0, 256, 0,
        pw_m2[l], msg_b2 + (size_t)l * 256, EE, 256, 256,
        agg32, nullptr, edges + 1, 2, 256, 256);
    k_cast_f2h<<<(unsigned)cdiv((long)NN * 256, 256), 256, 0, stream>>>(agg32, agg16, (long)NN * 256);
    // upd1: concat(h, agg) @ W1 + b1, relu -> tmp (z1)
    k_gemm<1, 0><<<grid(NN, 256), 256, 0, stream>>>(
        h16v, agg16, nullptr, nullptr, nullptr, 0, 256, 256,
        pw_u1[l], upd_b1 + (size_t)l * 256, NN, 256, 512,
        nullptr, z1, nullptr, 0, 256, 256);
    // upd2: tmp @ W2 + b2, residual into h32/h16
    k_gemm<0, 3><<<grid(NN, 256), 256, 0, stream>>>(
        z1, nullptr, nullptr, nullptr, nullptr, 0, 256, 0,
        pw_u2[l], upd_b2 + (size_t)l * 256, NN, 256, 256,
        h32, h16v, nullptr, 0, 256, 256);
  }

  // --- head ----------------------------------------------------------------
  k_rel<<<(unsigned)cdiv(QQ, 256), 256, 0, stream>>>(geom_x, qp, rel, QQ);
  // head1: concat(h[s], h[o], rel) @ W1 + b1, relu -> e16
  k_gemm<3, 0><<<grid(QQ, 256), 256, 0, stream>>>(
      h16v, nullptr, rel, qp, qp + 1, 2, 256, 32,
      pw_h1, head_b1, QQ, 256, 544, nullptr, e16, nullptr, 0, 256, 256);
  // head2: e16 @ W2 + b2 -> out [Q,26] fp32 (N padded to 32, masked store)
  k_gemm<0, 4><<<grid(QQ, 32), 256, 0, stream>>>(
      e16, nullptr, nullptr, nullptr, nullptr, 0, 256, 0,
      pw_h2, head_b2, QQ, 32, 256, out, nullptr, nullptr, 0, 26, 26);
}
